// FusedLlamaLowBitMultiDecoderlayer_63385127354808
// MI455X (gfx1250) — matrix operand order repaired and checked by reading, not compiled, not run on hardware
//
#include <hip/hip_runtime.h>
#include <cstdint>

// ---------------- Problem constants (from reference) ----------------
#define LYR   16
#define DMODEL 512
#define NH    4
#define DKH   128          // DMODEL / NH
#define HIDF  2048
#define KCONV 11
#define BB    8
#define TT    512
#define SS    512
#define MROWS (BB*TT)      // 4096
#define MSROW (BB*SS)      // 4096
#define ATT_SCALE 0.088388347648318447f   // 1/sqrt(128)

typedef __bf16 bf16;
typedef __attribute__((ext_vector_type(16))) __bf16 v16bf;
typedef __attribute__((ext_vector_type(8)))  float  v8f;
typedef __attribute__((ext_vector_type(4)))  int    v4i;

#define GLOBAL_AS __attribute__((address_space(1)))
#define LDS_AS    __attribute__((address_space(3)))

static __device__ __forceinline__ bf16 tobf(float v)   { return (bf16)v; }
static __device__ __forceinline__ bf16 tobf(int8_t v)  { return (bf16)(float)v; }
static __device__ __forceinline__ bf16 tobf(bf16 v)    { return v; }

// Async global->LDS (CDNA5 GLOBAL_LOAD_ASYNC_TO_LDS_B128, ASYNCcnt-tracked).
// Guarded: falls back to plain copies if the toolchain lacks the builtin.
#if defined(__has_builtin)
#if __has_builtin(__builtin_amdgcn_global_load_async_to_lds_b128)
#define HAVE_ASYNC_LDS 1
#endif
#endif

#ifdef HAVE_ASYNC_LDS
template<int OFF>
static __device__ __forceinline__ void async_g2l_b128(const void* g, void* l) {
    // inttoptr casts: AS(1) flat==global; AS(3) uses addr[31:0] (the LDS offset)
    GLOBAL_AS v4i* gp = (GLOBAL_AS v4i*)(uintptr_t)g;
    LDS_AS    v4i* lp = (LDS_AS    v4i*)(uintptr_t)l;
    __builtin_amdgcn_global_load_async_to_lds_b128(gp, lp, OFF, 0);
}
static __device__ __forceinline__ void async_wait0() {
#if __has_builtin(__builtin_amdgcn_s_wait_asynccnt)
    __builtin_amdgcn_s_wait_asynccnt(0);
#else
    asm volatile("s_wait_asynccnt 0" ::: "memory");
#endif
}
#endif

// =====================================================================
// Tiled WMMA GEMM:  C[M,N] = epilogue( A[M,K](bf16,row-major) x B )
//   TB      : int8_t (weights, per-out-channel scale) or bf16
//   KMAJOR  : B stored [K,N] (true) or [N,K] (false)
//   RELU/SCALE/BIAS/RES : epilogue options
// Block tile 128x64x64, 256 threads = 8 waves, wave tile 32x32
// (eight v_wmma_f32_16x16x32_bf16 per staged tile: 2 k-steps x 2x2).
// A-tile is loaded via async global->LDS DMA (overlaps with int8->bf16
// dequant/transpose of the B-tile through VALU), then s_wait_asynccnt.
// Requires: M%128==0, N%64==0, K%64==0 (true for all call sites).
// =====================================================================
template<typename TB, bool KMAJOR, bool RELU, bool SCALE, bool BIAS, bool RES>
__global__ __launch_bounds__(256)
void gemm_wmma(const bf16* __restrict__ A, int lda,
               const TB*  __restrict__ Bm, int ldb,
               const float* __restrict__ scale, const float* __restrict__ bias,
               float alpha,
               const float* __restrict__ Res,
               float* __restrict__ C, int ldc,
               int M, int N, int Kd,
               long strideA, long strideB,
               long cOffOuter, long cOffInner, int zdiv)
{
    constexpr int BM = 128, BN = 64, BK = 64;
    constexpr int LAS = BK + 8;   // LDS row stride for A (bf16 elems), 144B (16B aligned)
    constexpr int LBS = BN + 8;   // LDS row stride for B
    __shared__ alignas(16) bf16 Asl[BM * LAS];
    __shared__ alignas(16) bf16 Bsl[BK * LBS];

    const int tid  = threadIdx.x;
    const int lane = tid & 31;
    const int wave = tid >> 5;
    const int wm   = (wave >> 1) * 32;   // wave row offset in block tile
    const int wn   = (wave & 1)  * 32;   // wave col offset
    const int bm   = blockIdx.y * BM;
    const int bn   = blockIdx.x * BN;
    const int z    = blockIdx.z;

    const bf16* Az = A  + (long)z * strideA;
    const TB*   Bz = Bm + (long)z * strideB;
    const long  coff = (long)(z / zdiv) * cOffOuter + (long)(z % zdiv) * cOffInner;
    float*       Cz = C + coff;
    const float* Rz = RES ? (Res + coff) : nullptr;

    const v8f vzero = {0.f,0.f,0.f,0.f,0.f,0.f,0.f,0.f};
    v8f acc[2][2];
    acc[0][0]=vzero; acc[0][1]=vzero; acc[1][0]=vzero; acc[1][1]=vzero;

    // staging coordinates (fixed per thread)
    const int am  = tid >> 1;            // 0..127 : A row
    const int akp = (tid & 1) * 32;      // 0,32   : A col chunk (32 bf16 = 64B)

    for (int k0 = 0; k0 < Kd; k0 += BK) {
        // ---- stage A tile (128x64 bf16): async DMA straight into LDS ----
        {
            const bf16* src = Az + (long)(bm + am) * lda + k0 + akp;
            bf16* dst = &Asl[am * LAS + akp];
#ifdef HAVE_ASYNC_LDS
            async_g2l_b128<0 >(src, dst);
            async_g2l_b128<16>(src, dst);
            async_g2l_b128<32>(src, dst);
            async_g2l_b128<48>(src, dst);
#else
            #pragma unroll
            for (int j = 0; j < 32; ++j) dst[j] = src[j];
#endif
        }
        // ---- stage B tile into canonical [BK][BN] layout (overlaps DMA) ----
        if constexpr (KMAJOR) {
            const int kk = tid >> 2;            // 0..63
            const int np = (tid & 3) * 16;      // 0..48
            const TB* src = Bz + (long)(k0 + kk) * ldb + bn + np;
            bf16* dst = &Bsl[kk * LBS + np];
            #pragma unroll
            for (int j = 0; j < 16; ++j) dst[j] = tobf(src[j]);
        } else {
            const int n  = tid >> 2;            // 0..63 (output channel)
            const int kp = (tid & 3) * 16;      // 0..48
            const TB* src = Bz + (long)(bn + n) * ldb + k0 + kp;
            if (k0 + BK < Kd) __builtin_prefetch(src + BK, 0, 1);  // global_prefetch_b8
            #pragma unroll
            for (int j = 0; j < 16; ++j) Bsl[(kp + j) * LBS + n] = tobf(src[j]);
        }
#ifdef HAVE_ASYNC_LDS
        async_wait0();
#endif
        __syncthreads();

        // ---- fragments per documented CDNA5 WMMA VGPR layouts ----
        // A 16x32 bf16: lane -> M = lane&15, Kbase = (lane>>4)*8,
        //               element e -> K = Kbase + e + (e>=8 ? 8 : 0)
        // B 32x16 bf16: lane -> N = lane&15, Kbase = (lane>>4)*8,
        //               element e -> K = Kbase + e + (e>=8 ? 8 : 0)
        const int mrow = lane & 15;
        const int kb   = (lane >> 4) * 8;
        #pragma unroll
        for (int ks = 0; ks < BK; ks += 32) {
            v16bf afr[2], bfr[2];
            #pragma unroll
            for (int i = 0; i < 2; ++i) {
                const bf16* ap = &Asl[(wm + i * 16 + mrow) * LAS + ks];
                #pragma unroll
                for (int e = 0; e < 16; ++e) afr[i][e] = ap[kb + e + ((e >= 8) ? 8 : 0)];
            }
            #pragma unroll
            for (int i = 0; i < 2; ++i) {
                const bf16* bp = &Bsl[(ks + kb) * LBS + wn + i * 16 + mrow];
                #pragma unroll
                for (int e = 0; e < 16; ++e) bfr[i][e] = bp[(e + ((e >= 8) ? 8 : 0)) * LBS];
            }
            #pragma unroll
            for (int i = 0; i < 2; ++i)
                #pragma unroll
                for (int j = 0; j < 2; ++j)
                    acc[i][j] = __builtin_amdgcn_wmma_f32_16x16x32_bf16(
                        false, afr[i], false, bfr[j], (short)0, acc[i][j], false, false);
        }
        __syncthreads();
    }

    // ---- epilogue: C 16x16 f32 layout: lane -> N=lane&15, vgpr r -> M=r+8*(lane>>4)
    #pragma unroll
    for (int i = 0; i < 2; ++i) {
        #pragma unroll
        for (int j = 0; j < 2; ++j) {
            const int gn = bn + wn + j * 16 + (lane & 15);
            float sc, bs = 0.f;
            if constexpr (SCALE) sc = scale[gn]; else sc = alpha;
            if constexpr (BIAS)  bs = bias[gn];
            #pragma unroll
            for (int r = 0; r < 8; ++r) {
                const int gm = bm + wm + i * 16 + r + 8 * (lane >> 4);
                float v = acc[i][j][r] * sc + bs;
                if constexpr (RELU) v = fmaxf(v, 0.f);
                const long off = (long)gm * ldc + gn;
                if constexpr (RES) v += Rz[off];
                Cz[off] = v;
            }
        }
    }
}

// =====================================================================
// LayerNorm over last dim; block=256 per row; bf16 or f32 output
// =====================================================================
template<bool OUTBF>
__global__ __launch_bounds__(256)
void ln_kernel(const float* __restrict__ x, const float* __restrict__ w,
               const float* __restrict__ b, void* __restrict__ outv, int rowlen)
{
    const long row = blockIdx.x;
    const float* xr = x + row * (long)rowlen;
    float s = 0.f, sq = 0.f;
    for (int j = threadIdx.x; j < rowlen; j += 256) { float v = xr[j]; s += v; sq += v * v; }
    #pragma unroll
    for (int off = 16; off; off >>= 1) { s += __shfl_xor(s, off, 32); sq += __shfl_xor(sq, off, 32); }
    __shared__ float rs_[8], rq_[8];
    const int wave = threadIdx.x >> 5, lane = threadIdx.x & 31;
    if (lane == 0) { rs_[wave] = s; rq_[wave] = sq; }
    __syncthreads();
    if (threadIdx.x == 0) {
        float a = 0.f, c = 0.f;
        for (int i = 0; i < 8; ++i) { a += rs_[i]; c += rq_[i]; }
        rs_[0] = a; rq_[0] = c;
    }
    __syncthreads();
    const float mean = rs_[0] / rowlen;
    const float var  = rq_[0] / rowlen - mean * mean;
    const float inv  = rsqrtf(var + 1e-5f);
    for (int j = threadIdx.x; j < rowlen; j += 256) {
        const float v = (xr[j] - mean) * inv * w[j] + b[j];
        if constexpr (OUTBF) ((bf16*)outv)[row * rowlen + j] = (bf16)v;
        else                 ((float*)outv)[row * rowlen + j] = v;
    }
}

// =====================================================================
// FSMN: depthwise conv1d (K=11) over time + inner residual + mask,
// then outer residual add: xout = xin + ( conv(h*mask) + h*mask ) * mask
// =====================================================================
__global__ __launch_bounds__(256)
void fsmn_kernel(const float* __restrict__ h, const float* __restrict__ mask,
                 const float* __restrict__ w, const float* __restrict__ xin,
                 float* __restrict__ xout)
{
    const long idx = (long)blockIdx.x * 256 + threadIdx.x;
    if (idx >= (long)BB * TT * DMODEL) return;
    const int  d  = (int)(idx % DMODEL);
    const long bt = idx / DMODEL;
    const int  t  = (int)(bt % TT);
    const int  b  = (int)(bt / TT);
    const float mk = mask[b * TT + t];
    const float center = h[idx] * mk;
    float acc = 0.f;
    #pragma unroll
    for (int j = 0; j < KCONV; ++j) {
        const int tj = t + j - (KCONV - 1) / 2;
        if (tj >= 0 && tj < TT) {
            const float mj = mask[b * TT + tj];
            acc += w[d * KCONV + j] * h[((long)b * TT + tj) * DMODEL + d] * mj;
        }
    }
    xout[idx] = xin[idx] + (acc + center) * mk;
}

// =====================================================================
// Masked softmax over S=512; one wave per (b,h,t) row; writes bf16 probs
// =====================================================================
__global__ __launch_bounds__(256)
void softmax_kernel(const float* __restrict__ scores, const float* __restrict__ mmask,
                    bf16* __restrict__ probs)
{
    const int wave = threadIdx.x >> 5, lane = threadIdx.x & 31;
    const long row = (long)blockIdx.x * 8 + wave;       // over B*H*T rows
    const float* sr = scores + row * SS;
    const int b = (int)(row / ((long)NH * TT));
    const float* mr = mmask + (long)b * SS;
    float v[16];
    float mx = -3.4e38f;
    #pragma unroll
    for (int jj = 0; jj < 16; ++jj) {
        const int s0 = lane + jj * 32;
        const float sc = (mr[s0] > 0.f) ? sr[s0] : -1e9f;
        v[jj] = sc; mx = fmaxf(mx, sc);
    }
    #pragma unroll
    for (int off = 16; off; off >>= 1) mx = fmaxf(mx, __shfl_xor(mx, off, 32));
    float sum = 0.f;
    #pragma unroll
    for (int jj = 0; jj < 16; ++jj) {
        const int s0 = lane + jj * 32;
        const float p = (mr[s0] > 0.f) ? __expf(v[jj] - mx) : 0.f;
        v[jj] = p; sum += p;
    }
    #pragma unroll
    for (int off = 16; off; off >>= 1) sum += __shfl_xor(sum, off, 32);
    const float inv = 1.f / sum;
    #pragma unroll
    for (int jj = 0; jj < 16; ++jj)
        probs[row * SS + lane + jj * 32] = (bf16)(v[jj] * inv);
}

// =====================================================================
// Split-heads pack: src f32 [B, Tlen, rowLen] (+colOff) -> bf16 [B*H, Tlen, DKH]
// =====================================================================
__global__ __launch_bounds__(256)
void pack_heads(const float* __restrict__ src, int rowLen, int colOff, int Tlen,
                bf16* __restrict__ out)
{
    const long idx = (long)blockIdx.x * 256 + threadIdx.x;
    if (idx >= (long)BB * Tlen * DMODEL) return;
    const int  c = (int)(idx % DMODEL);
    const long bt = idx / DMODEL;
    const int  t = (int)(bt % Tlen);
    const int  b = (int)(bt / Tlen);
    const int  h = c / DKH, dk = c % DKH;
    out[(((long)(b * NH + h) * Tlen + t) * DKH) + dk] =
        (bf16)src[((long)b * Tlen + t) * rowLen + colOff + c];
}

__global__ __launch_bounds__(256)
void cvt_bf16(const float* __restrict__ src, bf16* __restrict__ dst, long n)
{
    const long idx = (long)blockIdx.x * 256 + threadIdx.x;
    if (idx < n) dst[idx] = (bf16)src[idx];
}

// =====================================================================
// Host orchestration
// =====================================================================
extern "C" void kernel_launch(void* const* d_in, const int* in_sizes, int n_in,
                              void* d_out, int out_size, void* d_ws, size_t ws_size,
                              hipStream_t stream)
{
    (void)in_sizes; (void)n_in; (void)out_size; (void)ws_size;
    // ---- inputs (setup_inputs dict order) ----
    const float*  x_in   = (const float*)d_in[0];
    const float*  tmask  = (const float*)d_in[1];
    const float*  memory = (const float*)d_in[2];
    const float*  mmask  = (const float*)d_in[3];
    const float*  ln0w = (const float*)d_in[4],  *ln0b = (const float*)d_in[5];
    const float*  ln1w = (const float*)d_in[6],  *ln1b = (const float*)d_in[7];
    const float*  ln2w = (const float*)d_in[8],  *ln2b = (const float*)d_in[9];
    const float*  fnw  = (const float*)d_in[10], *fnb  = (const float*)d_in[11];
    const int8_t* w1q  = (const int8_t*)d_in[12];
    const float*  w1s  = (const float*)d_in[13], *w1b = (const float*)d_in[14];
    const int8_t* w2q  = (const int8_t*)d_in[15];
    const float*  w2s  = (const float*)d_in[16];
    const int8_t* wqq  = (const int8_t*)d_in[17];
    const float*  wqs  = (const float*)d_in[18], *qb  = (const float*)d_in[19];
    const int8_t* wkvq = (const int8_t*)d_in[20];
    const float*  wkvs = (const float*)d_in[21], *kvbi = (const float*)d_in[22];
    const int8_t* woq  = (const int8_t*)d_in[23];
    const float*  wos  = (const float*)d_in[24], *outb = (const float*)d_in[25];
    const float*  fsw  = (const float*)d_in[26];

    // ---- workspace carve-up ----
    char* ws = (char*)d_ws;
    size_t off = 0;
    auto carve = [&](size_t bytes) { size_t r = off; off += (bytes + 255) & ~(size_t)255; return r; };
    const size_t SC_ELEMS = (size_t)BB * NH * TT * SS;   // 8.4M elems
    float* h1     = (float*)(ws + carve(SC_ELEMS * 4));                 // M*HID f32 / scores f32 (shared)
    bf16*  Abuf   = (bf16*) (ws + carve(SC_ELEMS * 2));                 // M*HID bf16 / probs bf16 (shared)
    float* h2     = (float*)(ws + carve((size_t)MROWS * DMODEL * 4));   // gemm2 out / q f32
    float* h3     = (float*)(ws + carve((size_t)MROWS * DMODEL * 4));   // ln1 out / attn f32
    float* kvf    = (float*)(ws + carve((size_t)MSROW * 2 * DMODEL * 4));
    bf16*  Kb     = (bf16*) (ws + carve((size_t)BB * NH * SS * DKH * 2));
    bf16*  Vb     = (bf16*) (ws + carve((size_t)BB * NH * SS * DKH * 2));
    bf16*  Qb     = (bf16*) (ws + carve((size_t)BB * NH * TT * DKH * 2));
    bf16*  membf  = (bf16*) (ws + carve((size_t)MSROW * DMODEL * 2));
    bf16*  attnb  = (bf16*) (ws + carve((size_t)MROWS * DMODEL * 2));
    float* scoresf = h1;
    bf16*  probs   = Abuf;
    float* xb = (float*)d_out;   // running activation lives in d_out

    // init: x -> d_out, memory -> bf16
    (void)hipMemcpyAsync(xb, x_in, (size_t)MROWS * DMODEL * sizeof(float),
                         hipMemcpyDeviceToDevice, stream);
    {
        const long n = (long)MSROW * DMODEL;
        cvt_bf16<<<(unsigned)((n + 255) / 256), 256, 0, stream>>>(memory, membf, n);
    }

    const long elemsMD = (long)MROWS * DMODEL;
    const dim3 blk(256);

    for (int i = 0; i < LYR; ++i) {
        const int8_t* W1  = w1q  + (long)i * HIDF * DMODEL;
        const int8_t* W2  = w2q  + (long)i * DMODEL * HIDF;
        const int8_t* Wq  = wqq  + (long)i * DMODEL * DMODEL;
        const int8_t* Wkv = wkvq + (long)i * 2 * DMODEL * DMODEL;
        const int8_t* Wo  = woq  + (long)i * DMODEL * DMODEL;

        // ---------- FFN block ----------
        ln_kernel<true><<<MROWS, blk, 0, stream>>>(xb, ln0w + i*DMODEL, ln0b + i*DMODEL, Abuf, DMODEL);

        gemm_wmma<int8_t,false,true,true,true,false>
            <<<dim3(HIDF/64, MROWS/128, 1), blk, 0, stream>>>(
            Abuf, DMODEL, W1, DMODEL, w1s + (long)i*HIDF, w1b + (long)i*HIDF, 1.f,
            nullptr, h1, HIDF, MROWS, HIDF, DMODEL, 0, 0, 0, 0, 1);

        ln_kernel<true><<<MROWS, blk, 0, stream>>>(h1, fnw + (long)i*HIDF, fnb + (long)i*HIDF, Abuf, HIDF);

        gemm_wmma<int8_t,false,false,true,false,false>
            <<<dim3(DMODEL/64, MROWS/128, 1), blk, 0, stream>>>(
            Abuf, HIDF, W2, HIDF, w2s + (long)i*DMODEL, nullptr, 1.f,
            nullptr, h2, DMODEL, MROWS, DMODEL, HIDF, 0, 0, 0, 0, 1);

        ln_kernel<false><<<MROWS, blk, 0, stream>>>(h2, ln1w + i*DMODEL, ln1b + i*DMODEL, h3, DMODEL);

        fsmn_kernel<<<(unsigned)((elemsMD + 255) / 256), blk, 0, stream>>>(
            h3, tmask, fsw + (long)i * DMODEL * KCONV, xb, xb);

        // ---------- cross-attention block ----------
        ln_kernel<true><<<MROWS, blk, 0, stream>>>(xb, ln2w + i*DMODEL, ln2b + i*DMODEL, Abuf, DMODEL);

        gemm_wmma<int8_t,false,false,true,true,false>
            <<<dim3(DMODEL/64, MROWS/128, 1), blk, 0, stream>>>(
            Abuf, DMODEL, Wq, DMODEL, wqs + (long)i*DMODEL, qb + (long)i*DMODEL, 1.f,
            nullptr, h2, DMODEL, MROWS, DMODEL, DMODEL, 0, 0, 0, 0, 1);

        gemm_wmma<int8_t,false,false,true,true,false>
            <<<dim3(2*DMODEL/64, MSROW/128, 1), blk, 0, stream>>>(
            membf, DMODEL, Wkv, DMODEL, wkvs + (long)i*2*DMODEL, kvbi + (long)i*2*DMODEL, 1.f,
            nullptr, kvf, 2*DMODEL, MSROW, 2*DMODEL, DMODEL, 0, 0, 0, 0, 1);

        pack_heads<<<(unsigned)((elemsMD + 255) / 256), blk, 0, stream>>>(h2,  DMODEL,   0,      TT, Qb);
        pack_heads<<<(unsigned)((elemsMD + 255) / 256), blk, 0, stream>>>(kvf, 2*DMODEL, 0,      SS, Kb);
        pack_heads<<<(unsigned)((elemsMD + 255) / 256), blk, 0, stream>>>(kvf, 2*DMODEL, DMODEL, SS, Vb);

        // scores[z, T, S] = alpha * Q[z] @ K[z]^T   (z = b*H + h)
        gemm_wmma<bf16,false,false,false,false,false>
            <<<dim3(SS/64, TT/128, BB*NH), blk, 0, stream>>>(
            Qb, DKH, Kb, DKH, nullptr, nullptr, ATT_SCALE,
            nullptr, scoresf, SS, TT, SS, DKH,
            (long)TT*DKH, (long)SS*DKH, (long)TT*SS, 0, 1);

        softmax_kernel<<<(unsigned)(BB*NH*TT/8), blk, 0, stream>>>(scoresf, mmask, probs);

        // attn[b, T, h*DKH : (h+1)*DKH] = P[z] @ V[z]   (V is [K=S, N=DKH] -> KMAJOR)
        gemm_wmma<bf16,true,false,false,false,false>
            <<<dim3(DKH/64, TT/128, BB*NH), blk, 0, stream>>>(
            probs, SS, Vb, DKH, nullptr, nullptr, 1.f,
            nullptr, h3, DMODEL, TT, DKH, SS,
            (long)TT*SS, (long)SS*DKH, (long)TT*DMODEL, DKH, NH);

        cvt_bf16<<<(unsigned)((elemsMD + 255) / 256), blk, 0, stream>>>(h3, attnb, elemsMD);

        // x = x + attn @ Wo^T * s + bo   (residual fused in epilogue)
        gemm_wmma<int8_t,false,false,true,true,true>
            <<<dim3(DMODEL/64, MROWS/128, 1), blk, 0, stream>>>(
            attnb, DMODEL, Wo, DMODEL, wos + (long)i*DMODEL, outb + (long)i*DMODEL, 1.f,
            xb, xb, DMODEL, MROWS, DMODEL, DMODEL, 0, 0, 0, 0, 1);
    }
}
